// Full_Graph_5420248727996
// MI455X (gfx1250) — matrix operand order repaired and checked by reading, compiled, not run on hardware
//
#include <hip/hip_runtime.h>
#include <math.h>

#define Bn 8
#define Cc 10
#define Hh 128
#define Ww 128
#define HW (Hh*Ww)
#define CHW (Cc*HW)

typedef __bf16 bf16x16 __attribute__((ext_vector_type(16)));
typedef float  f32x8   __attribute__((ext_vector_type(8)));

// f32 -> bf16 round-to-nearest-even
__device__ __forceinline__ unsigned short f2bf(float f) {
    unsigned u = __builtin_bit_cast(unsigned, f);
    unsigned r = (u + 0x7FFFu + ((u >> 16) & 1u)) >> 16;
    return (unsigned short)r;
}
__device__ __forceinline__ __bf16 us2bf(unsigned short v) {
    return __builtin_bit_cast(__bf16, v);
}
__device__ __forceinline__ float sigmoidf_(float x) {
    return 1.0f / (1.0f + __expf(-x));
}

// Compile-time A-operand LDS element offset for the conv kernel (lo lane half).
// K mapping: 16-block t, slot s9 = t*8 + (e&7); lo half = channels 0..4,
// hi half = channels 5..9 (constant +1950 element delta). Pad slots -> zero row 30
// (hi: row 45). Fully constant-folded inside unrolled loops.
__host__ __device__ constexpr int conv_aoff(int s, int e) {
    int t  = 2 * s + (e < 8 ? 0 : 1);
    int s9 = t * 8 + (e & 7);
    if (s9 >= 45) return 30 * 130;           // zero pad row (lo); hi adds 1950 -> row 45
    int ci = s9 / 9, tap = s9 % 9;
    return (ci * 3 + tap / 3) * 130 + (tap % 3);
}

// ---------------------------------------------------------------------------
// Kernel 1: comp_att = sigmoid(1x1 conv over cat[h0,h1], 2C -> 1)
// ---------------------------------------------------------------------------
__global__ void __launch_bounds__(256) att_kernel(
    const float* __restrict__ h0, const float* __restrict__ h1,
    const float* __restrict__ w, const float* __restrict__ bias,
    float* __restrict__ att) {
    int p = blockIdx.x * 256 + threadIdx.x;
    if (p >= Bn * HW) return;
    int bb = p / HW;
    int yx = p % HW;
    const float* p0 = h0 + bb * CHW + yx;
    const float* p1 = h1 + bb * CHW + yx;
    float acc = bias[0];
#pragma unroll
    for (int i = 0; i < Cc; ++i) acc += w[i] * p0[i * HW];
#pragma unroll
    for (int i = 0; i < Cc; ++i) acc += w[Cc + i] * p1[i * HW];
    att[p] = sigmoidf_(acc);
}

// ---------------------------------------------------------------------------
// Kernel 2: comp_h = relu(BN(conv3x3(cat[f,h0*att]))) + relu(BN(conv3x3(cat[f,h1*att])))
// Implicit GEMM via v_wmma_f32_16x16x32_bf16; shared f-part accumulator.
// One block = (batch, row); 4 waves x 2 tiles of 16 pixels cover the 128-wide row.
// A gathers are single ds_load_u16 with compile-time immediate offsets.
// ---------------------------------------------------------------------------
__global__ void __launch_bounds__(128) conv_wmma_kernel(
    const float* __restrict__ f, const float* __restrict__ h0,
    const float* __restrict__ h1, const float* __restrict__ att,
    const float* __restrict__ rel_w,
    const float* __restrict__ bn_g, const float* __restrict__ bn_b,
    const float* __restrict__ bn_m, const float* __restrict__ bn_v,
    float* __restrict__ comp_h) {
    __shared__ unsigned short sF [46 * 130];   // rows 0..29 data, rows 30/45 zero pad
    __shared__ unsigned short sC0[46 * 130];
    __shared__ unsigned short sC1[46 * 130];
    __shared__ unsigned short sW [2 * 96 * 16]; // weights [part][K][n], K-mapped
    __shared__ float sAtt[3 * 130];

    const int tid = threadIdx.x;
    const int bb  = blockIdx.x >> 7;
    const int y   = blockIdx.x & 127;

    // ---- phase 1: att halo rows, mapped weight matrix, zero pad rows ----
    for (int i = tid; i < 3 * 130; i += 128) {
        int row = i / 130, xl = i % 130;
        int gy = y + row - 1, gx = xl - 1;
        float v = 0.f;
        if (gy >= 0 && gy < Hh && gx >= 0 && gx < Ww)
            v = att[bb * HW + gy * Ww + gx];
        sAtt[i] = v;
    }
    for (int i = tid; i < 2 * 96 * 16; i += 128) {
        int part = i / (96 * 16);
        int r = i % (96 * 16);
        int k = r >> 4, n = r & 15;
        int t = k >> 4, u = k & 15;
        int half = u >> 3, up = u & 7;
        int s9 = t * 8 + up;
        float v = 0.f;
        if (s9 < 45 && n < Cc) {
            int ci = s9 / 9 + half * 5;
            int tap = s9 % 9, dy = tap / 3, dx = tap % 3;
            v = rel_w[((n * (2 * Cc) + part * Cc + ci) * 3 + dy) * 3 + dx];
        }
        sW[i] = f2bf(v);
    }
    for (int i = tid; i < 130; i += 128) {      // zero pad rows 30 and 45
        sF [3900 + i] = 0; sF [5850 + i] = 0;
        sC0[3900 + i] = 0; sC0[5850 + i] = 0;
        sC1[3900 + i] = 0; sC1[5850 + i] = 0;
    }
    __syncthreads();

    // ---- phase 2: stage input halo rows as bf16 (rows 0..29: (ci*3+row)*130+x) ----
    for (int i = tid; i < Cc * 3 * 130; i += 128) {
        int ci = i / 390;
        int r2 = i % 390;
        int row = r2 / 130, xl = r2 % 130;
        int gy = y + row - 1, gx = xl - 1;
        float vf = 0.f, v0 = 0.f, v1 = 0.f;
        if (gy >= 0 && gy < Hh && gx >= 0 && gx < Ww) {
            int gidx = bb * CHW + ci * HW + gy * Ww + gx;
            float a = sAtt[row * 130 + xl];
            vf = f[gidx];
            v0 = h0[gidx] * a;
            v1 = h1[gidx] * a;
        }
        sF[i]  = f2bf(vf);
        sC0[i] = f2bf(v0);
        sC1[i] = f2bf(v1);
    }
    __syncthreads();

    const int lane   = tid & 31;
    const int wave   = tid >> 5;
    const int laneHi = lane >> 4;
    const int n      = lane & 15;

    // B fragments: table order, K = 32s + 8*hi + e (e<8), 32s + 16 + 8*hi + (e-8) (e>=8)
    bf16x16 Bw[2][3];
#pragma unroll
    for (int part = 0; part < 2; ++part)
#pragma unroll
        for (int s = 0; s < 3; ++s) {
            bf16x16 bfrag;
#pragma unroll
            for (int e = 0; e < 16; ++e) {
                int K = 32 * s + (laneHi ? 8 : 0) + e + (e < 8 ? 0 : 8);
                bfrag[e] = us2bf(sW[part * 96 * 16 + K * 16 + n]);
            }
            Bw[part][s] = bfrag;
        }

    float bnm = 0.f, bnb = 0.f, bninv = 1.f;
    if (n < Cc) {
        bnm = bn_m[n];
        bnb = bn_b[n];
        bninv = bn_g[n] / sqrtf(bn_v[n] + 1e-5f);
    }

#pragma unroll
    for (int tile = 0; tile < 2; ++tile) {
        const int x0  = wave * 32 + tile * 16;
        const int xmh = x0 + (lane & 15) + laneHi * 1950;   // per-lane base

        f32x8 accF = {0.f, 0.f, 0.f, 0.f, 0.f, 0.f, 0.f, 0.f};
#pragma unroll
        for (int s = 0; s < 3; ++s) {
            bf16x16 a;
#pragma unroll
            for (int e = 0; e < 16; ++e) a[e] = us2bf(sF[conv_aoff(s, e) + xmh]);
            accF = __builtin_amdgcn_wmma_f32_16x16x32_bf16(
                false, a, false, Bw[0][s], (short)0, accF, false, false);
        }
        f32x8 acc0 = accF;
#pragma unroll
        for (int s = 0; s < 3; ++s) {
            bf16x16 a;
#pragma unroll
            for (int e = 0; e < 16; ++e) a[e] = us2bf(sC0[conv_aoff(s, e) + xmh]);
            acc0 = __builtin_amdgcn_wmma_f32_16x16x32_bf16(
                false, a, false, Bw[1][s], (short)0, acc0, false, false);
        }
        f32x8 acc1 = accF;
#pragma unroll
        for (int s = 0; s < 3; ++s) {
            bf16x16 a;
#pragma unroll
            for (int e = 0; e < 16; ++e) a[e] = us2bf(sC1[conv_aoff(s, e) + xmh]);
            acc1 = __builtin_amdgcn_wmma_f32_16x16x32_bf16(
                false, a, false, Bw[1][s], (short)0, acc1, false, false);
        }

        if (n < Cc) {   // D: lane l, reg r -> M = r + 8*laneHi, N = l&15
            float* outp = comp_h + bb * CHW + n * HW + y * Ww + x0 + (laneHi ? 8 : 0);
#pragma unroll
            for (int r = 0; r < 8; ++r) {
                float y0 = (acc0[r] - bnm) * bninv + bnb;
                float y1 = (acc1[r] - bnm) * bninv + bnb;
                outp[r] = fmaxf(y0, 0.f) + fmaxf(y1, 0.f);
            }
        }
    }
}

// ---------------------------------------------------------------------------
// Kernel 3: ConvGRU via WMMA.
//   gates: [16px x 32K] x [32 x 32] (two 16-wide N tiles), K = [comp_h(10); f(10); 0(12)]
//   cand : [16px x 32K] x [32 x 16],                       K = [comp_h(10); reset*f(10); 0]
// One block = (batch, row); 4 waves x 2 tiles of 16 pixels.
// ---------------------------------------------------------------------------
__global__ void __launch_bounds__(128) gru_wmma_kernel(
    const float* __restrict__ f, const float* __restrict__ comp_h,
    const float* __restrict__ gw, const float* __restrict__ gb,
    const float* __restrict__ cw, const float* __restrict__ cb,
    float* __restrict__ out) {
    __shared__ unsigned short sA [32 * 128];  // bf16 A panel: rows=K, cols=x
    __shared__ unsigned short sGW[32 * 32];   // gates B [K][n]
    __shared__ unsigned short sCW[32 * 16];   // cand  B [K][n]
    __shared__ float sG [20 * 128];           // sigmoided gates (f32)
    __shared__ float sFf[10 * 128];           // f (f32) for the GRU blend
    __shared__ float sGB[20];
    __shared__ float sCB[10];

    const int tid = threadIdx.x;
    const int bb  = blockIdx.x >> 7;
    const int y   = blockIdx.x & 127;

    // ---- phase 1: stage A panel (comp_h, f), weights, biases ----
    for (int i = tid; i < Cc * 128; i += 128) {
        int ch = i >> 7, x = i & 127;
        int gidx = bb * CHW + ch * HW + y * Ww + x;
        float cv = comp_h[gidx];
        float fv = f[gidx];
        sA[ch * 128 + x]        = f2bf(cv);
        sA[(Cc + ch) * 128 + x] = f2bf(fv);
        sFf[ch * 128 + x]       = fv;
    }
    for (int i = tid; i < 12 * 128; i += 128) sA[20 * 128 + i] = 0;  // K pad rows
    for (int i = tid; i < 32 * 32; i += 128) {
        int k = i >> 5, n = i & 31;
        sGW[i] = (k < 2 * Cc && n < 2 * Cc) ? f2bf(gw[n * (2 * Cc) + k]) : (unsigned short)0;
    }
    for (int i = tid; i < 32 * 16; i += 128) {
        int k = i >> 4, n = i & 15;
        sCW[i] = (k < 2 * Cc && n < Cc) ? f2bf(cw[n * (2 * Cc) + k]) : (unsigned short)0;
    }
    if (tid < 20) sGB[tid] = gb[tid];
    if (tid < 10) sCB[tid] = cb[tid];
    __syncthreads();

    const int lane   = tid & 31;
    const int wave   = tid >> 5;
    const int laneHi = lane >> 4;
    const int n      = lane & 15;
    const int aBase  = laneHi * 1024;  // hi lane half: K += 8  (8 rows * 128)

    // B fragments (table order: K = hi*8 + e + (e<8 ? 0 : 8), single k-step)
    bf16x16 Bg[2], Bc;
#pragma unroll
    for (int j = 0; j < 2; ++j) {
        bf16x16 bfrag;
#pragma unroll
        for (int e = 0; e < 16; ++e) {
            int K = (laneHi ? 8 : 0) + e + (e < 8 ? 0 : 8);
            bfrag[e] = us2bf(sGW[K * 32 + j * 16 + n]);
        }
        Bg[j] = bfrag;
    }
#pragma unroll
    for (int e = 0; e < 16; ++e) {
        int K = (laneHi ? 8 : 0) + e + (e < 8 ? 0 : 8);
        Bc[e] = us2bf(sCW[K * 16 + n]);
    }

    // ---- phase 2: gates WMMA, sigmoid, spill to sG ----
#pragma unroll
    for (int tile = 0; tile < 2; ++tile) {
        const int x0 = wave * 32 + tile * 16;
        const int x  = x0 + (lane & 15);
        bf16x16 a;
#pragma unroll
        for (int e = 0; e < 16; ++e) {
            int Klo = e + (e < 8 ? 0 : 8);
            a[e] = us2bf(sA[Klo * 128 + x + aBase]);
        }
#pragma unroll
        for (int j = 0; j < 2; ++j) {
            f32x8 acc = {0.f, 0.f, 0.f, 0.f, 0.f, 0.f, 0.f, 0.f};
            acc = __builtin_amdgcn_wmma_f32_16x16x32_bf16(
                false, a, false, Bg[j], (short)0, acc, false, false);
            int gch = j * 16 + n;
            if (gch < 2 * Cc) {
                float gbv = sGB[gch];
#pragma unroll
                for (int r = 0; r < 8; ++r)
                    sG[gch * 128 + x0 + r + laneHi * 8] = sigmoidf_(acc[r] + gbv);
            }
        }
    }
    __syncthreads();

    // ---- phase 3: overwrite A rows 10..19 with reset*f ----
    for (int i = tid; i < Cc * 128; i += 128) {
        int ch = i >> 7, x = i & 127;
        sA[(Cc + ch) * 128 + x] = f2bf(sG[ch * 128 + x] * sFf[ch * 128 + x]);
    }
    __syncthreads();

    // ---- phase 4: candidate WMMA, tanh, GRU blend, store ----
#pragma unroll
    for (int tile = 0; tile < 2; ++tile) {
        const int x0 = wave * 32 + tile * 16;
        const int x  = x0 + (lane & 15);
        bf16x16 a;
#pragma unroll
        for (int e = 0; e < 16; ++e) {
            int Klo = e + (e < 8 ? 0 : 8);
            a[e] = us2bf(sA[Klo * 128 + x + aBase]);
        }
        f32x8 acc = {0.f, 0.f, 0.f, 0.f, 0.f, 0.f, 0.f, 0.f};
        acc = __builtin_amdgcn_wmma_f32_16x16x32_bf16(
            false, a, false, Bc, (short)0, acc, false, false);
        if (n < Cc) {
            float cbv = sCB[n];
            float* outp = out + bb * CHW + n * HW + y * Ww;
#pragma unroll
            for (int r = 0; r < 8; ++r) {
                int xp = x0 + r + laneHi * 8;
                float cand = tanhf(acc[r] + cbv);
                float u    = sG[(Cc + n) * 128 + xp];
                float fo   = sFf[n * 128 + xp];
                outp[xp] = (1.0f - u) * fo + u * cand;
            }
        }
    }
}

// ---------------------------------------------------------------------------
extern "C" void kernel_launch(void* const* d_in, const int* in_sizes, int n_in,
                              void* d_out, int out_size, void* d_ws, size_t ws_size,
                              hipStream_t stream) {
    const float* f     = (const float*)d_in[0];
    const float* h0    = (const float*)d_in[1];
    const float* h1    = (const float*)d_in[2];
    // d_in[3] (p_nodes) and d_in[4] (xf) are unused by the reference
    const float* att_w = (const float*)d_in[5];
    const float* att_b = (const float*)d_in[6];
    const float* rel_w = (const float*)d_in[7];
    const float* bn_g  = (const float*)d_in[8];
    const float* bn_b  = (const float*)d_in[9];
    const float* bn_m  = (const float*)d_in[10];
    const float* bn_v  = (const float*)d_in[11];
    const float* gw    = (const float*)d_in[12];
    const float* gb    = (const float*)d_in[13];
    const float* cw    = (const float*)d_in[14];
    const float* cb    = (const float*)d_in[15];

    float* out_f   = (float*)d_out;           // f_node_new: B*C*H*W
    float* out_att = out_f + Bn * CHW;        // comp_att:  B*1*H*W
    float* comp_h  = (float*)d_ws;            // scratch:   B*C*H*W floats

    att_kernel<<<(Bn * HW + 255) / 256, 256, 0, stream>>>(h0, h1, att_w, att_b, out_att);
    conv_wmma_kernel<<<Bn * Hh, 128, 0, stream>>>(f, h0, h1, out_att, rel_w,
                                                  bn_g, bn_b, bn_m, bn_v, comp_h);
    gru_wmma_kernel<<<Bn * Hh, 128, 0, stream>>>(f, comp_h, gw, gb, cw, cb, out_f);
}
